// TargetAwareCrossAttention_86955907875120
// MI455X (gfx1250) — matrix operand order repaired and checked by reading, not compiled, not run on hardware
//
#include <hip/hip_runtime.h>
#include <stdint.h>

// Problem constants (from reference)
#define B_   128
#define T_   32
#define S_   2048
#define D_   128
#define H_   4
#define DH_  32
#define ST_  32                 // S-chunk handled per loop iteration
#define NCHUNK (S_ / ST_)
#define LDC_ 136                // bf16 buffers: 128 cols + 8 halfword pad (16B)
#define LDV_ 40                 // bf16 buffers: 32 cols + 8 halfword pad
#define LDF_ 132                // f32 TDM-landed buffers: 128 cols + 4 dword pad (16B)

typedef __bf16 bf16_t;
typedef __attribute__((ext_vector_type(16))) __bf16    v16bf;
typedef __attribute__((ext_vector_type(8)))  __bf16    v8bf;
typedef __attribute__((ext_vector_type(8)))  float     v8f;
typedef __attribute__((ext_vector_type(4)))  uint32_t  u32x4;
typedef __attribute__((ext_vector_type(8)))  uint32_t  u32x8;

struct SMem {
  bf16_t Q[T_ * LDC_];         // (Q + bq)/scale bf16 [t][d]; reused as attn A-operand in epilogue
  bf16_t Cs[ST_ * LDC_];       // bf16-converted activation tile (target / context chunk)
  bf16_t Ks[ST_ * LDC_];       // K chunk, [s][d]   (B-operand layout for Q.K^T)
  bf16_t Vt[D_ * LDV_];        // V chunk transposed, [d][s] (B-operand layout for S.V)
  bf16_t Ss[H_ * T_ * LDV_];   // scores bf16, [h][t][s]
  float  Cf[2][ST_ * LDF_];    // TDM-landed f32 tiles, padded rows (132 dwords)
};

static __device__ inline v8f wmma_bf16(v16bf a, v16bf b, v8f c) {
  return __builtin_amdgcn_wmma_f32_16x16x32_bf16(false, a, false, b, (short)0, c,
                                                 false, false);
}

// ---- Tensor Data Mover: 1-D tile of f32, global -> LDS, with LDS row padding
// (pad 4 DWORDs after every 128 DWORDs -> 132-dword row stride in LDS).
static __device__ inline void tdm_load_f32(uint32_t lds_off, const float* gptr,
                                           uint32_t tensor_elems, uint32_t tile_elems) {
  const uint64_t ga = (uint64_t)(uintptr_t)gptr;
  u32x4 g0;
  g0[0] = 1u;                                                // count=1, user D#
  g0[1] = __builtin_amdgcn_readfirstlane(lds_off);           // lds_addr (bytes)
  g0[2] = __builtin_amdgcn_readfirstlane((uint32_t)ga);      // global_addr[31:0]
  g0[3] = (__builtin_amdgcn_readfirstlane((uint32_t)(ga >> 32)) & 0x01FFFFFFu)
          | 0x80000000u;                                     // global_addr[56:32] | type=2
  u32x8 g1;
  g1[0] = 0x00020000u                                        // data_size = 4 bytes
        | (1u << 20)                                         // pad_enable
        | (6u << 22)                                         // pad_interval: 128 DWORDs
        | (3u << 25);                                        // pad_amount: 4 DWORDs
  g1[1] = (tensor_elems & 0xFFFFu) << 16;                    // tensor_dim0[15:0]
  g1[2] = (tensor_elems >> 16) & 0xFFFFu;                    // tensor_dim0[31:16]
  g1[3] = (tile_elems & 0xFFFFu) << 16;                      // tile_dim0
  g1[4] = 0;                                                 // tile_dim1/2 unused (1-D)
  g1[5] = tensor_elems;                                      // tensor_dim0_stride[31:0]
  g1[6] = 0;
  g1[7] = 0;
  asm volatile("tensor_load_to_lds %0, %1" :: "s"(g0), "s"(g1) : "memory");
}

// Cooperative convert pass: 32x128 f32 tile (LDS, ld=LDF_) -> bf16 tile (LDS,
// ld=LDC_). 256 threads, 16 contiguous elements each: 4x ds_load_b128 +
// 8x v_cvt_pk_bf16_f32 + 2x ds_store_b128 per thread.
static __device__ inline void cvt_tile(const float* __restrict__ src,
                                       bf16_t* __restrict__ dst, int tid) {
  const int row = tid >> 3;            // 8 threads per 128-col row
  const int col = (tid & 7) * 16;
  const float4* p = reinterpret_cast<const float4*>(src + row * LDF_ + col);
  v8bf* d = reinterpret_cast<v8bf*>(dst + row * LDC_ + col);
  float4 x0 = p[0], x1 = p[1], x2 = p[2], x3 = p[3];
  v8bf lo, hi;
  lo[0] = (bf16_t)x0.x; lo[1] = (bf16_t)x0.y; lo[2] = (bf16_t)x0.z; lo[3] = (bf16_t)x0.w;
  lo[4] = (bf16_t)x1.x; lo[5] = (bf16_t)x1.y; lo[6] = (bf16_t)x1.z; lo[7] = (bf16_t)x1.w;
  hi[0] = (bf16_t)x2.x; hi[1] = (bf16_t)x2.y; hi[2] = (bf16_t)x2.z; hi[3] = (bf16_t)x2.w;
  hi[4] = (bf16_t)x3.x; hi[5] = (bf16_t)x3.y; hi[6] = (bf16_t)x3.z; hi[7] = (bf16_t)x3.w;
  d[0] = lo;
  d[1] = hi;
}

// A fragment (16x32 bf16) from LDS row-major [m][k].
static __device__ inline v16bf load_a(const bf16_t* p0, int ld, int row0, int k0, int lane) {
  const int r = lane & 15, g = lane >> 4;
  const bf16_t* p = p0 + (row0 + r) * ld + k0 + g * 16;
  v16bf a;
#pragma unroll
  for (int i = 0; i < 16; ++i) a[i] = p[i];
  return a;
}

// B fragment (32x16 bf16) from LDS where the operand is stored [n][k] row-major.
static __device__ inline v16bf load_b(const bf16_t* p0, int ld, int col0, int k0, int lane) {
  const int n = lane & 15, kh = (lane >> 4) * 16;
  const bf16_t* p = p0 + (col0 + n) * ld + k0 + kh;
  v16bf b;
#pragma unroll
  for (int i = 0; i < 16; ++i) b[i] = p[i];
  return b;
}

// B fragment straight from a global f32 weight matrix W (row-major [dout][din]);
// nn.Linear computes x @ W^T, so W itself is already the [n][k] operand layout.
static __device__ inline v16bf load_b_wf32(const float* W, int col0, int k0, int lane) {
  const int n = lane & 15, kh = (lane >> 4) * 16;
  const float4* p = reinterpret_cast<const float4*>(W + (size_t)(col0 + n) * D_ + k0 + kh);
  v16bf b;
#pragma unroll
  for (int j = 0; j < 4; ++j) {
    float4 v = p[j];
    b[j * 4 + 0] = (bf16_t)v.x;
    b[j * 4 + 1] = (bf16_t)v.y;
    b[j * 4 + 2] = (bf16_t)v.z;
    b[j * 4 + 3] = (bf16_t)v.w;
  }
  return b;
}

__global__ __launch_bounds__(256)
void fused_xattn_kernel(const float* __restrict__ target,
                        const float* __restrict__ context,
                        const int*   __restrict__ mask,
                        const float* __restrict__ Wq, const float* __restrict__ bq,
                        const float* __restrict__ Wk, const float* __restrict__ bk,
                        const float* __restrict__ Wv, const float* __restrict__ bv,
                        const float* __restrict__ Wo, const float* __restrict__ bo,
                        const float* __restrict__ scale,
                        float* __restrict__ out) {
  __shared__ SMem sm;
  const int b    = blockIdx.x;
  const int tid  = threadIdx.x;
  const int lane = tid & 31;
  const int w    = tid >> 5;                 // wave id, 0..7
  const float inv_scale = 1.0f / scale[0];

  const uint32_t cf_off[2] = {
    (uint32_t)(uintptr_t)(void*)&sm.Cf[0][0],
    (uint32_t)(uintptr_t)(void*)&sm.Cf[1][0]
  };
  const float* ctx_base = context + (size_t)b * S_ * D_;

  // ---------------- Prologue ----------------
  // TDM: target tile (32x128 f32) -> Cf[0]; overlap with weight-fragment loads.
  if (w == 0)
    tdm_load_f32(cf_off[0], target + (size_t)b * T_ * D_, T_ * D_, T_ * D_);

  // Resident Wk/Wv B-fragments: waves 0-3 own K projection, waves 4-7 own V.
  const bool isV = (w >= 4);
  const int  proj_nb = w & 3;                // output-column block (32 cols)
  v16bf wfrag[2][4];                         // [n-tile][k-step]
  {
    const float* W = isV ? Wv : Wk;
#pragma unroll
    for (int nt = 0; nt < 2; ++nt)
#pragma unroll
      for (int ks = 0; ks < 4; ++ks)
        wfrag[nt][ks] = load_b_wf32(W, proj_nb * 32 + nt * 16, ks * 32, lane);
  }

  if (w == 0) __builtin_amdgcn_s_wait_tensorcnt(0);
  __syncthreads();
  cvt_tile(sm.Cf[0], sm.Cs, tid);            // target f32 -> bf16
  __syncthreads();

  // Q = (target @ Wq^T + bq) * inv_scale -> bf16 LDS
  {
    const int mt = w & 1, nb = w >> 1;       // 2 m-tiles x 4 n-blocks across 8 waves
    const int row0 = mt * 16;
#pragma unroll
    for (int nt = 0; nt < 2; ++nt) {
      const int col0 = nb * 32 + nt * 16;
      v8f c = {};
#pragma unroll
      for (int ks = 0; ks < 4; ++ks) {
        v16bf a  = load_a(sm.Cs, LDC_, row0, ks * 32, lane);
        v16bf bb = load_b_wf32(Wq, col0, ks * 32, lane);
        c = wmma_bf16(a, bb, c);
      }
      const int n = lane & 15, m0g = (lane >> 4) * 8;
      const int col = col0 + n;
      const float bias = bq[col];
#pragma unroll
      for (int i = 0; i < 8; ++i)
        sm.Q[(row0 + m0g + i) * LDC_ + col] = (bf16_t)((c[i] + bias) * inv_scale);
    }
  }
  __syncthreads();

  // Per-wave attention role: head h, t-tile t0; Q A-fragment is loop-invariant.
  const int h  = w >> 1;
  const int t0 = (w & 1) * 16;
  const v16bf qfrag = load_a(sm.Q, LDC_, t0, h * DH_, lane);

  // Kick off double-buffered TDM pipeline: chunks 0 and 1 in flight.
  if (w == 0) {
    tdm_load_f32(cf_off[0], ctx_base + 0 * (ST_ * D_), S_ * D_, ST_ * D_);
    tdm_load_f32(cf_off[1], ctx_base + 1 * (ST_ * D_), S_ * D_, ST_ * D_);
  }

  v8f acc0 = {}, acc1 = {};                  // out tiles (t0, dh 0..15 / 16..31)
  const int* mrow = mask + (size_t)b * S_;

  for (int sc = 0; sc < NCHUNK; ++sc) {
    const int buf = sc & 1;

    // Chunk sc has landed once <=1 transfer remains in flight (TDM is in-order).
    if (w == 0) {
      if (sc < NCHUNK - 2) __builtin_amdgcn_s_wait_tensorcnt(1);
      else                 __builtin_amdgcn_s_wait_tensorcnt(0);
    }
    __syncthreads();

    // Phase 1: convert f32 chunk -> bf16 staging (one cvt per element).
    cvt_tile(sm.Cf[buf], sm.Cs, tid);
    __syncthreads();

    // Cf[buf] is free again: refill it with chunk sc+2; DMA overlaps all the
    // WMMA work below.
    if (w == 0 && sc + 2 < NCHUNK)
      tdm_load_f32(cf_off[buf], ctx_base + (size_t)(sc + 2) * (ST_ * D_),
                   S_ * D_, ST_ * D_);

    // Phase 2: K/V chunk projections (ctx @ W^T + bias), weights in registers.
#pragma unroll
    for (int mt2 = 0; mt2 < 2; ++mt2) {
      const int row0 = mt2 * 16;
      v16bf a[4];
#pragma unroll
      for (int ks = 0; ks < 4; ++ks) a[ks] = load_a(sm.Cs, LDC_, row0, ks * 32, lane);
#pragma unroll
      for (int nt = 0; nt < 2; ++nt) {
        const int col0 = proj_nb * 32 + nt * 16;
        v8f c = {};
#pragma unroll
        for (int ks = 0; ks < 4; ++ks) c = wmma_bf16(a[ks], wfrag[nt][ks], c);
        const int n = lane & 15, m0g = (lane >> 4) * 8;
        const int col = col0 + n;
        const float bias = isV ? bv[col] : bk[col];
        if (!isV) {
#pragma unroll
          for (int i = 0; i < 8; ++i)
            sm.Ks[(row0 + m0g + i) * LDC_ + col] = (bf16_t)(c[i] + bias);
        } else {
#pragma unroll
          for (int i = 0; i < 8; ++i)
            sm.Vt[col * LDV_ + row0 + m0g + i] = (bf16_t)(c[i] + bias);
        }
      }
    }
    __syncthreads();

    // Phase 3: scores = Qscaled . K^T, mask, -> bf16 Ss[h][t][s]
#pragma unroll
    for (int st2 = 0; st2 < 2; ++st2) {
      const int s0 = st2 * 16;
      v16bf kb = load_b(sm.Ks, LDC_, s0, h * DH_, lane);
      v8f c = {};
      c = wmma_bf16(qfrag, kb, c);
      const int n = lane & 15, m0g = (lane >> 4) * 8;
      const int sl = s0 + n;
      const bool live = (mrow[sc * ST_ + sl] != 0);
      bf16_t* srow = sm.Ss + (h * T_) * LDV_;
#pragma unroll
      for (int i = 0; i < 8; ++i) {
        float v = live ? c[i] : -__builtin_inff();
        srow[(t0 + m0g + i) * LDV_ + sl] = (bf16_t)v;
      }
    }
    __syncthreads();

    // Phase 4: out += scores . V  (accumulators live across the whole S loop)
    {
      v16bf sfrag = load_a(sm.Ss + (h * T_) * LDV_, LDV_, t0, 0, lane);
      v16bf vb0 = load_b(sm.Vt, LDV_, h * DH_ + 0,  0, lane);
      v16bf vb1 = load_b(sm.Vt, LDV_, h * DH_ + 16, 0, lane);
      acc0 = wmma_bf16(sfrag, vb0, acc0);
      acc1 = wmma_bf16(sfrag, vb1, acc1);
    }
  }

  // ---------------- Epilogue: stage attn (bf16) into sm.Q, out = attn @ Wo^T + bo
  {
    const int n = lane & 15, m0g = (lane >> 4) * 8;
#pragma unroll
    for (int i = 0; i < 8; ++i) {
      sm.Q[(t0 + m0g + i) * LDC_ + h * DH_ + 0  + n] = (bf16_t)acc0[i];
      sm.Q[(t0 + m0g + i) * LDC_ + h * DH_ + 16 + n] = (bf16_t)acc1[i];
    }
  }
  __syncthreads();
  {
    const int mt = w & 1, nb = w >> 1;
    const int row0 = mt * 16;
#pragma unroll
    for (int nt = 0; nt < 2; ++nt) {
      const int col0 = nb * 32 + nt * 16;
      v8f c = {};
#pragma unroll
      for (int ks = 0; ks < 4; ++ks) {
        v16bf a  = load_a(sm.Q, LDC_, row0, ks * 32, lane);
        v16bf bb = load_b_wf32(Wo, col0, ks * 32, lane);
        c = wmma_bf16(a, bb, c);
      }
      const int n = lane & 15, m0g = (lane >> 4) * 8;
      const int col = col0 + n;
      const float bias = bo[col];
#pragma unroll
      for (int i = 0; i < 8; ++i)
        out[((size_t)b * T_ + row0 + m0g + i) * D_ + col] = c[i] + bias;
    }
  }
}

extern "C" void kernel_launch(void* const* d_in, const int* in_sizes, int n_in,
                              void* d_out, int out_size, void* d_ws, size_t ws_size,
                              hipStream_t stream) {
  (void)in_sizes; (void)n_in; (void)out_size; (void)d_ws; (void)ws_size;
  const float* target  = (const float*)d_in[0];
  const float* context = (const float*)d_in[1];
  const int*   mask    = (const int*)  d_in[2];
  const float* Wq = (const float*)d_in[3];
  const float* bq = (const float*)d_in[4];
  const float* Wk = (const float*)d_in[5];
  const float* bk = (const float*)d_in[6];
  const float* Wv = (const float*)d_in[7];
  const float* bv = (const float*)d_in[8];
  const float* Wo = (const float*)d_in[9];
  const float* bo = (const float*)d_in[10];
  const float* scale = (const float*)d_in[11];
  float* out = (float*)d_out;

  fused_xattn_kernel<<<dim3(B_), dim3(256), 0, stream>>>(
      target, context, mask, Wq, bq, Wk, bk, Wv, bv, Wo, bo, scale, out);
}
